// GNOBlock_63797444215460
// MI455X (gfx1250) — matrix operand order repaired and checked by reading, not compiled, not run on hardware
//
#include <hip/hip_runtime.h>
#include <hip/hip_bf16.h>

// ---------------------------------------------------------------------------
// GNO block for MI455X (gfx1250, wave32, WMMA).
// Heavy path: batched 4-layer MLP as v_wmma_f32_16x16x32_bf16 chains,
// one wave per query (16 padded neighbor slots == one M-tile), k-reduction
// fused into the last-layer epilogue. Weight fragments stored nt-major so the
// kt-inner loop streams contiguous 1KB fragments.
// ---------------------------------------------------------------------------

typedef __attribute__((ext_vector_type(16))) __bf16 v16bf;
typedef __attribute__((ext_vector_type(8)))  float  v8f;

union BF16Frag {
    uint4          u[2];
    v16bf          v;
    unsigned short s[16];
};

__device__ __forceinline__ unsigned short f2bf(float f) {
    union { float f; unsigned u; } v; v.f = f;
    unsigned r = v.u + 0x7FFFu + ((v.u >> 16) & 1u);   // round-to-nearest-even
    return (unsigned short)(r >> 16);
}

__device__ __forceinline__ float gelu_tanh(float x) {
    float x3 = x * x * x;
    return 0.5f * x * (1.0f + tanhf(0.7978845608f * (x + 0.044715f * x3)));
}

// ---------------------------------------------------------------------------
// 1) sinusoidal embedding -> bf16 [npts, 192]; layout per coord d:
//    [d*64 + f] = sin, [d*64 + 32 + f] = cos,  f in [0,32)
// ---------------------------------------------------------------------------
__global__ void embed_kernel(const float* __restrict__ pts,
                             unsigned short* __restrict__ emb, int npts) {
    int gid = blockIdx.x * blockDim.x + threadIdx.x;
    if (gid >= npts * 96) return;
    int p = gid / 96, rdf = gid % 96;
    int d = rdf / 32, f = rdf % 32;
    float c    = pts[p * 3 + d];
    float freq = __expf(-0.28782313662425572f * (float)f); // 1/10000^(f/32)
    float ang  = c * freq;
    emb[p * 192 + d * 64 + f]      = f2bf(__sinf(ang));
    emb[p * 192 + d * 64 + 32 + f] = f2bf(__cosf(ang));
}

// ---------------------------------------------------------------------------
// 2) pack fp32 weights into per-lane WMMA B-fragment order (bf16).
//    B (32x16) per wave32 lane: col = lane&15, h = lane>>4,
//    element e (0..15) -> k_local = (e < 8) ? 8*h + e : 16 + 8*h + (e - 8).
//    Fragment index f = nt*kT + kt  (nt-major: kt-inner loop is contiguous),
//    storage: wpack[off + f*512 + lane*16 + e].
// ---------------------------------------------------------------------------
__global__ void pack_kernel(const float* __restrict__ W1, const float* __restrict__ W2,
                            const float* __restrict__ W3, const float* __restrict__ W4,
                            unsigned short* __restrict__ wpack) {
    int gid = blockIdx.x * blockDim.x + threadIdx.x;
    if (gid >= 122880) return;
    const float* W; int N, K, off;
    if      (gid <  49152) { W = W1; N = 128; K = 384; off = 0;      }
    else if (gid <  81920) { W = W2; N = 256; K = 128; off = 49152;  }
    else if (gid < 114688) { W = W3; N = 128; K = 256; off = 81920;  }
    else                   { W = W4; N =  64; K = 128; off = 114688; }
    int r    = gid - off;
    int frag = r >> 9;
    int q    = r & 511;
    int lane = q >> 4;
    int e    = q & 15;
    int kT   = K >> 5;
    int nt   = frag / kT;            // nt-major storage
    int kt   = frag - nt * kT;
    int h    = lane >> 4;
    int k    = kt * 32 + ((e < 8) ? (8 * h + e) : (16 + 8 * h + (e - 8)));
    int n    = nt * 16 + (lane & 15);
    wpack[gid] = f2bf(W[k * N + n]);
}

// ---------------------------------------------------------------------------
// 3) brute-force radius search, LDS-tiled. 12 nearest within R, padded to 16.
// ---------------------------------------------------------------------------
#define RAD2 0.0036f

__global__ void neigh_kernel(const float* __restrict__ y, const float* __restrict__ x,
                             int* __restrict__ idxp, int* __restrict__ cnt,
                             int n, int m) {
    __shared__ float ty[256 * 3];
    int q = blockIdx.x * 256 + threadIdx.x;
    float qx = 0.f, qy = 0.f, qz = 0.f;
    if (q < m) { qx = x[q * 3]; qy = x[q * 3 + 1]; qz = x[q * 3 + 2]; }
    float bd[12]; int bi[12];
#pragma unroll
    for (int s = 0; s < 12; ++s) { bd[s] = __builtin_inff(); bi[s] = 0; }

    for (int t0 = 0; t0 < n; t0 += 256) {
        int src = t0 + (int)threadIdx.x;
        if (src < n) {
            ty[threadIdx.x * 3 + 0] = y[src * 3 + 0];
            ty[threadIdx.x * 3 + 1] = y[src * 3 + 1];
            ty[threadIdx.x * 3 + 2] = y[src * 3 + 2];
        } else {
            ty[threadIdx.x * 3 + 0] = 1e30f;
            ty[threadIdx.x * 3 + 1] = 1e30f;
            ty[threadIdx.x * 3 + 2] = 1e30f;
        }
        __syncthreads();
        int lim = n - t0; if (lim > 256) lim = 256;
        for (int j = 0; j < lim; ++j) {
            float dx = qx - ty[j * 3 + 0];
            float dy = qy - ty[j * 3 + 1];
            float dz = qz - ty[j * 3 + 2];
            float d2 = dx * dx + dy * dy + dz * dz;
            if (d2 <= RAD2 && d2 < bd[11]) {          // rare path (~0.1%)
                int cand = t0 + j;
#pragma unroll
                for (int s = 0; s < 12; ++s) {        // bubble insert (sorted asc)
                    if (d2 < bd[s]) {
                        float td = bd[s]; int ti = bi[s];
                        bd[s] = d2; bi[s] = cand;
                        d2 = td; cand = ti;
                    }
                }
            }
        }
        __syncthreads();
    }
    if (q < m) {
        int c = 0;
#pragma unroll
        for (int s = 0; s < 12; ++s) if (bd[s] <= RAD2) ++c;
#pragma unroll
        for (int s = 0; s < 16; ++s)
            idxp[q * 16 + s] = (s < 12 && bd[s] <= RAD2) ? bi[s] : 0;
        cnt[q] = c;
    }
}

// ---------------------------------------------------------------------------
// 4) fused MLP + k-reduction. One wave32 per query; 2 waves per block.
//    LDS ping-pong buffers, stride-padded for conflict-free ds_load_b128.
// ---------------------------------------------------------------------------
#define NW     2          // waves per block
#define FSTR   392        // feat buffer row stride (ushorts), 784B = 16B-aligned
#define HSTR   264        // h buffer row stride (ushorts),   528B = 16B-aligned
#define OFF_L1 0
#define OFF_L2 49152
#define OFF_L3 81920
#define OFF_L4 114688

template <int WIN, int WOUT, bool GELU>
__device__ __forceinline__ void mlp_layer(const unsigned short* __restrict__ wpack,
                                          unsigned fragBase,
                                          const float* __restrict__ bias,
                                          const unsigned short* inL, int inStride,
                                          unsigned short* outL, int outStride,
                                          int lane) {
    constexpr int nT = WOUT >> 4;
    constexpr int kT = WIN >> 5;
    const int r = lane & 15, h = lane >> 4;
    __builtin_prefetch(wpack + fragBase, 0, 0);       // global_prefetch_b8
    for (int nt = 0; nt < nT; ++nt) {
        float bval = bias[nt * 16 + r];               // col == lane&15 in C layout
        v8f acc = { bval, bval, bval, bval, bval, bval, bval, bval };
#pragma unroll
        for (int kt = 0; kt < kT; ++kt) {
            BF16Frag a, b;
            // A 16x32 bf16 (ISA layout): row = lane&15; lanes<16: K 0-7,16-23;
            // lanes>=16: K 8-15,24-31  -> two ds_load_b128
            const uint4* ap = (const uint4*)(inL + r * inStride + kt * 32 + 8 * h);
            a.u[0] = ap[0];
            a.u[1] = ap[2];                           // +16 ushorts = +32B
            // B fragment: 16 contiguous bf16 per lane; nt-major storage makes
            // the kt loop stream contiguous 1KB fragments
            const uint4* bp = (const uint4*)(wpack + fragBase +
                                             (unsigned)(nt * kT + kt) * 512u + lane * 16);
            b.u[0] = bp[0];
            b.u[1] = bp[1];
            acc = __builtin_amdgcn_wmma_f32_16x16x32_bf16(
                      false, a.v, false, b.v, (short)0, acc, false, false);
        }
        // C/D layout: VGPR v, lanes<16 -> row v, lanes>=16 -> row v+8; col = lane&15
#pragma unroll
        for (int v = 0; v < 8; ++v) {
            float xv = acc[v];
            if (GELU) xv = gelu_tanh(xv);
            outL[(v + 8 * h) * outStride + nt * 16 + r] = f2bf(xv);
        }
    }
}

__global__ void __launch_bounds__(NW * 32)
gno_mlp_kernel(const unsigned short* __restrict__ y_emb,
               const unsigned short* __restrict__ x_emb,
               const unsigned short* __restrict__ wpack,
               const int* __restrict__ idxp, const int* __restrict__ cnt,
               const float* __restrict__ f_y,
               const float* __restrict__ b1, const float* __restrict__ b2,
               const float* __restrict__ b3, const float* __restrict__ b4,
               float* __restrict__ out, int m) {
    __shared__ __align__(16) unsigned short s_feat[NW * 16 * FSTR];
    __shared__ __align__(16) unsigned short s_h[NW * 16 * HSTR];

    const int lane = (int)threadIdx.x & 31;
    const int wave = (int)threadIdx.x >> 5;
    const int q    = blockIdx.x * NW + wave;
    if (q >= m) return;                                // wave-uniform exit

    unsigned short* feat = s_feat + wave * 16 * FSTR;
    unsigned short* hbuf = s_h    + wave * 16 * HSTR;
    const int r = lane & 15, h = lane >> 4;
    const int cq = cnt[q];

    // ---- gather feat[16][384] = [y_emb[idx[row]] (192) | x_emb[q] (192)] ----
    {
        const uint4* xsrc = (const uint4*)(x_emb + (size_t)q * 192);
        for (int row = 0; row < 16; ++row) {
            int id = idxp[q * 16 + row];
            const uint4* ysrc = (const uint4*)(y_emb + (size_t)id * 192);
            uint4* dst = (uint4*)(feat + row * FSTR);  // FSTR%8==0 -> 16B aligned
            for (int s = lane; s < 48; s += 32) {
                if (s < 24) dst[s] = ysrc[s];
                else        dst[s] = xsrc[s - 24];
            }
        }
    }
    // same-wave LDS ops retire in order: stores visible to following ds loads

    // ---- layers 1..3 (WMMA + bias-in-accumulator + GELU, LDS ping-pong) ----
    mlp_layer<384, 128, true>(wpack, OFF_L1, b1, feat, FSTR, hbuf, HSTR, lane);
    mlp_layer<128, 256, true>(wpack, OFF_L2, b2, hbuf, HSTR, feat, FSTR, lane);
    mlp_layer<256, 128, true>(wpack, OFF_L3, b3, feat, FSTR, hbuf, HSTR, lane);

    // ---- layer 4 (128 -> 64) + f_y gather + mask + k-reduction in regs ----
    int   nidx[8];
    float msk[8];
#pragma unroll
    for (int v = 0; v < 8; ++v) {
        int row = v + 8 * h;
        nidx[v] = idxp[q * 16 + row];
        msk[v]  = (row < cq) ? 1.0f : 0.0f;
    }
    for (int nt = 0; nt < 4; ++nt) {
        float bval = b4[nt * 16 + r];
        v8f acc = { bval, bval, bval, bval, bval, bval, bval, bval };
#pragma unroll
        for (int kt = 0; kt < 4; ++kt) {
            BF16Frag a, b;
            const uint4* ap = (const uint4*)(hbuf + r * HSTR + kt * 32 + 8 * h);
            a.u[0] = ap[0];
            a.u[1] = ap[2];
            const uint4* bp = (const uint4*)(wpack + OFF_L4 +
                                             (unsigned)(nt * 4 + kt) * 512u + lane * 16);
            b.u[0] = bp[0];
            b.u[1] = bp[1];
            acc = __builtin_amdgcn_wmma_f32_16x16x32_bf16(
                      false, a.v, false, b.v, (short)0, acc, false, false);
        }
        float part = 0.0f;
#pragma unroll
        for (int v = 0; v < 8; ++v) {
            float fv = f_y[(size_t)nidx[v] * 64 + nt * 16 + r];
            part += acc[v] * fv * msk[v];
        }
        part += __shfl_xor(part, 16, 32);              // combine row halves
        if (h == 0) out[(size_t)q * 64 + nt * 16 + r] = part;
    }
}

// ---------------------------------------------------------------------------
// launch
// ---------------------------------------------------------------------------
extern "C" void kernel_launch(void* const* d_in, const int* in_sizes, int n_in,
                              void* d_out, int out_size, void* d_ws, size_t ws_size,
                              hipStream_t stream) {
    const float* y   = (const float*)d_in[0];
    const float* x   = (const float*)d_in[1];
    const float* f_y = (const float*)d_in[2];
    const float* W1  = (const float*)d_in[3];
    const float* b1  = (const float*)d_in[4];
    const float* W2  = (const float*)d_in[5];
    const float* b2  = (const float*)d_in[6];
    const float* W3  = (const float*)d_in[7];
    const float* b3  = (const float*)d_in[8];
    const float* W4  = (const float*)d_in[9];
    const float* b4  = (const float*)d_in[10];

    const int n = in_sizes[0] / 3;
    const int m = in_sizes[1] / 3;

    // workspace layout (256B aligned slabs)
    char*  ws  = (char*)d_ws;
    size_t o   = 0;
    auto   alloc = [&](size_t bytes) { size_t c = o; o += (bytes + 255) & ~(size_t)255; return c; };
    unsigned short* ws_yemb  = (unsigned short*)(ws + alloc((size_t)n * 192 * 2));
    unsigned short* ws_xemb  = (unsigned short*)(ws + alloc((size_t)m * 192 * 2));
    unsigned short* ws_wpack = (unsigned short*)(ws + alloc((size_t)122880 * 2));
    int*            ws_idxp  = (int*)(ws + alloc((size_t)m * 16 * 4));
    int*            ws_cnt   = (int*)(ws + alloc((size_t)m * 4));
    (void)ws_size; (void)n_in; (void)out_size;

    embed_kernel<<<(n * 96 + 255) / 256, 256, 0, stream>>>(y, ws_yemb, n);
    embed_kernel<<<(m * 96 + 255) / 256, 256, 0, stream>>>(x, ws_xemb, m);
    pack_kernel<<<(122880 + 255) / 256, 256, 0, stream>>>(W1, W2, W3, W4, ws_wpack);
    neigh_kernel<<<(m + 255) / 256, 256, 0, stream>>>(y, x, ws_idxp, ws_cnt, n, m);
    gno_mlp_kernel<<<(m + NW - 1) / NW, NW * 32, 0, stream>>>(
        ws_yemb, ws_xemb, ws_wpack, ws_idxp, ws_cnt, f_y,
        b1, b2, b3, b4, (float*)d_out, m);
}
